// EdgeGNN_43087111914331
// MI455X (gfx1250) — matrix operand order repaired and checked by reading, not compiled, not run on hardware
//
#include <hip/hip_runtime.h>
#include <hip/hip_bf16.h>

typedef __attribute__((ext_vector_type(16))) _Float16 v16h;
typedef __attribute__((ext_vector_type(8)))  _Float16 v8h;
typedef __attribute__((ext_vector_type(8)))  float    v8f;

#define NN 50000
#define EE 800000
#define HH 128
#define PP 100000
#define OUTC 86
#define OUTPAD 96

// ---------------------------------------------------------------------------
// Utility kernels
// ---------------------------------------------------------------------------
__global__ void fill_kernel(float* __restrict__ p, float v, long long n) {
    long long i = (long long)blockIdx.x * blockDim.x + threadIdx.x;
    if (i < n) p[i] = v;
}

__global__ void f32_to_f16_kernel(const float* __restrict__ in, _Float16* __restrict__ out,
                                  long long n) {
    long long i = (long long)blockIdx.x * blockDim.x + threadIdx.x;
    if (i < n) out[i] = (_Float16)in[i];
}

// degree accumulate: deg[dst[e]] += 1
__global__ void deg_kernel(float* __restrict__ deg, const int* __restrict__ dst, int e_cnt) {
    int e = blockIdx.x * blockDim.x + threadIdx.x;
    if (e < e_cnt) atomicAdd(&deg[dst[e]], 1.0f);
}

__global__ void dinv_kernel(float* __restrict__ dinv, const float* __restrict__ deg, int n) {
    int i = blockIdx.x * blockDim.x + threadIdx.x;
    if (i < n) dinv[i] = rsqrtf(deg[i] + 1.0f);
}

// ---------------------------------------------------------------------------
// Pack a row-major f32 weight [Ksrc, Nsrc] into WMMA B-operand layout (f16):
// Bp[((tn*nKb + kb)*32 + lane)*16 + e]
//   lanes 0..15  hold K = k0 + {0..7, 16..23} (e<8 -> k0+e, e>=8 -> k0+16+(e-8)), col = tn*16 + (lane&15)
//   lanes 16..31 hold K = k0 + {8..15, 24..31}
// Zero-pad columns >= Nsrc (used for OUT=86 -> 96 padding).
// ---------------------------------------------------------------------------
__global__ void pack_b_kernel(const float* __restrict__ B, _Float16* __restrict__ out,
                              int Ksrc, int Nsrc, int Npad) {
    const int nKb = Ksrc >> 5;
    const int nTN = Npad >> 4;
    long long total = (long long)nTN * nKb * 512;
    long long idx = (long long)blockIdx.x * blockDim.x + threadIdx.x;
    if (idx >= total) return;
    int e    = (int)(idx & 15);
    int lane = (int)((idx >> 4) & 31);
    int r    = (int)(idx >> 9);
    int kb   = r % nKb;
    int tn   = r / nKb;
    int k    = (kb << 5) + ((e >> 3) << 4) + ((lane >> 4) << 3) + (e & 7);
    int n    = (tn << 4) + (lane & 15);
    float v  = (n < Nsrc) ? B[(size_t)k * Nsrc + n] : 0.0f;
    out[idx] = (_Float16)v;
}

// ---------------------------------------------------------------------------
// Row-block WMMA GEMM with async-staged B:
//  * whole packed B is copied Global->LDS once per workgroup with
//    global_load_async_to_lds_b128 (ASYNCcnt), then s_wait_asynccnt + barrier
//  * one wave owns a 16-row block and ALL NTN column tiles; the A fragment is
//    loaded once per K-step and reused across NTN WMMAs
//  * per K-step, ALL NTN B fragments are loaded from LDS into registers first
//    (batched ds_load_b128 issue, staged s_wait_dscnt), then the WMMA chain runs
//  * f32 accumulation, fused bias/ReLU, optional f16 downconvert
// Dynamic shared memory is the only LDS allocation -> base offset 0.
// ---------------------------------------------------------------------------
template <int NTN, bool RELU, bool OUT_HALF>
__global__ void wmma_gemm_rows_kernel(const _Float16* __restrict__ A,
                                      const _Float16* __restrict__ Bp,
                                      const float* __restrict__ bias,
                                      float* __restrict__ outF,
                                      _Float16* __restrict__ outH,
                                      int M, int K, int ldOut, int maxCol) {
    extern __shared__ _Float16 bsh[];
    const int nKb = K >> 5;

    // ---- cooperative async copy of packed B into LDS (16B chunks) ---------
    {
        const int chunks = (NTN * nKb * 1024) >> 4;   // bytes/16
        for (int c = threadIdx.x; c < chunks; c += blockDim.x) {
            unsigned int       ldsOff = (unsigned int)c << 4;          // LDS byte addr
            unsigned long long gaddr  = (unsigned long long)(Bp + ((size_t)c << 3));
            asm volatile("global_load_async_to_lds_b128 %0, %1, off"
                         :: "v"(ldsOff), "v"(gaddr) : "memory");
        }
        asm volatile("s_wait_asynccnt 0x0" ::: "memory");
    }
    __syncthreads();

    const int wave = threadIdx.x >> 5;
    const int lane = threadIdx.x & 31;
    const int rowTiles = M >> 4;
    int t = blockIdx.x * (blockDim.x >> 5) + wave;    // wave-uniform -> EXEC all 1s
    if (t < rowTiles) {
        const int hlf = lane >> 4;
        const int lr  = lane & 15;
        const _Float16* aRow = A + (size_t)((t << 4) + lr) * K;

        v8f zero = {};
        v8f acc[NTN];
#pragma unroll
        for (int tn = 0; tn < NTN; ++tn) acc[tn] = zero;

        for (int kb = 0; kb < nKb; ++kb) {
            const int k0 = kb << 5;
            // A 16x32 f16 layout: lane<16 -> K {k0..+7, k0+16..+23}; lane>=16 shifted by 8
            v8h alo = *(const v8h*)(aRow + k0 + (hlf << 3));
            v8h ahi = *(const v8h*)(aRow + k0 + 16 + (hlf << 3));
            v16h a  = __builtin_shufflevector(alo, ahi, 0, 1, 2, 3, 4, 5, 6, 7,
                                                         8, 9, 10, 11, 12, 13, 14, 15);
            if (kb + 1 < nKb)
                __builtin_prefetch(aRow + ((kb + 1) << 5), 0, 0);  // global_prefetch_b8

            // batch-issue all NTN B-fragment loads from LDS, then run WMMA chain
            const _Float16* bk = bsh + ((size_t)((kb << 5) + lane) << 4);
            v16h bf[NTN];
#pragma unroll
            for (int tn = 0; tn < NTN; ++tn)
                bf[tn] = *(const v16h*)(bk + ((size_t)(tn * nKb) << 9));  // ds_load_b128 x2
#pragma unroll
            for (int tn = 0; tn < NTN; ++tn)
                acc[tn] = __builtin_amdgcn_wmma_f32_16x16x32_f16(false, a, false, bf[tn],
                                                                 (short)0, acc[tn],
                                                                 false, false);
        }

        // C/D layout: lane<16 -> rows t*16+0..7, col=lr ; lane>=16 -> rows t*16+8..15
        const int row0 = (t << 4) + (hlf << 3);
#pragma unroll
        for (int tn = 0; tn < NTN; ++tn) {
            const int col = (tn << 4) + lr;
            float bv = 0.0f;
            if (bias && col < maxCol) bv = bias[col];
#pragma unroll
            for (int i = 0; i < 8; ++i) {
                float v = acc[tn][i] + bv;
                if (RELU) v = fmaxf(v, 0.0f);
                if (OUT_HALF) {
                    outH[(size_t)(row0 + i) * ldOut + col] = (_Float16)v;
                } else {
                    if (col < maxCol) outF[(size_t)(row0 + i) * ldOut + col] = v;
                }
            }
        }
    }
}

// ---------------------------------------------------------------------------
// GCN edge aggregation: agg[dst] += hw[src] * dinv[src]*dinv[dst]
// one thread per (edge, 4-feature chunk); float4 load + 4 global f32 atomics
// agg is 25.6 MB -> L2-resident on MI455X (192 MB L2), atomics land in L2.
// ---------------------------------------------------------------------------
__global__ void edge_agg_kernel(const float* __restrict__ hw, float* __restrict__ agg,
                                const int* __restrict__ src, const int* __restrict__ dst,
                                const float* __restrict__ dinv, int e_cnt) {
    long long tid = (long long)blockIdx.x * blockDim.x + threadIdx.x;
    if (tid >= (long long)e_cnt * 32) return;
    int e  = (int)(tid >> 5);
    int c4 = ((int)tid & 31) << 2;
    int s = src[e], d = dst[e];
    float coef = dinv[s] * dinv[d];
    const float4 v = *(const float4*)(hw + (size_t)s * HH + c4);
    float* ap = agg + (size_t)d * HH + c4;
    atomicAdd(ap + 0, v.x * coef);
    atomicAdd(ap + 1, v.y * coef);
    atomicAdd(ap + 2, v.z * coef);
    atomicAdd(ap + 3, v.w * coef);
}

// self-loop contribution + GCN bias: agg[i][k] += hw[i][k]*dinv[i]^2 + b[k]
__global__ void selfloop_bias_kernel(float* __restrict__ agg, const float* __restrict__ hw,
                                     const float* __restrict__ dinv,
                                     const float* __restrict__ b, long long total) {
    long long idx = (long long)blockIdx.x * blockDim.x + threadIdx.x;
    if (idx >= total) return;
    int i = (int)(idx >> 7);
    int k = (int)(idx & 127);
    float di = dinv[i];
    agg[idx] += hw[idx] * di * di + b[k];
}

// ---------------------------------------------------------------------------
// BatchNorm stats, coalesced two-phase.
// ---------------------------------------------------------------------------
__global__ void bn_partial_kernel(const float* __restrict__ agg,
                                  float* __restrict__ colsum, float* __restrict__ colsq,
                                  int n) {
    __shared__ float s1[256];
    __shared__ float s2[256];
    const int c   = threadIdx.x & 127;
    const int hlf = threadIdx.x >> 7;     // 0 or 1
    const int r0  = blockIdx.x * 64;
    float s = 0.0f, q = 0.0f;
    for (int r = hlf; r < 64; r += 2) {
        int row = r0 + r;
        if (row < n) {
            float v = agg[(size_t)row * HH + c];
            s += v;
            q += v * v;
        }
    }
    s1[threadIdx.x] = s;
    s2[threadIdx.x] = q;
    __syncthreads();
    if (hlf == 0) {
        atomicAdd(&colsum[c], s1[threadIdx.x] + s1[threadIdx.x + 128]);
        atomicAdd(&colsq[c],  s2[threadIdx.x] + s2[threadIdx.x + 128]);
    }
}

__global__ void bn_finalize_kernel(const float* __restrict__ colsum,
                                   const float* __restrict__ colsq,
                                   float* __restrict__ mean, float* __restrict__ rstd,
                                   int n) {
    int k = threadIdx.x;   // 128 threads
    float m   = colsum[k] / n;
    float var = colsq[k] / n - m * m;
    mean[k] = m;
    rstd[k] = rsqrtf(var + 1e-5f);
}

// BN affine + ReLU, output downconverted to f16 for the next WMMA stage
__global__ void bn_apply_kernel(const float* __restrict__ agg, const float* __restrict__ gamma,
                                const float* __restrict__ beta, const float* __restrict__ mean,
                                const float* __restrict__ rstd, _Float16* __restrict__ out,
                                long long total) {
    long long idx = (long long)blockIdx.x * blockDim.x + threadIdx.x;
    if (idx >= total) return;
    int k = (int)(idx & 127);
    float v = gamma[k] * (agg[idx] - mean[k]) * rstd[k] + beta[k];
    out[idx] = (_Float16)fmaxf(v, 0.0f);
}

// gather node pair features -> combined[P, 256] f16 (concat h[a], h[b])
__global__ void gather_pairs_kernel(const _Float16* __restrict__ h2,
                                    const int* __restrict__ pairs,
                                    _Float16* __restrict__ comb, int p_cnt) {
    long long tid = (long long)blockIdx.x * blockDim.x + threadIdx.x;
    if (tid >= (long long)p_cnt * 64) return;
    int p = (int)(tid >> 6);
    int q = (int)tid & 63;                 // 64 chunks of 4 halves = 256 cols
    int node = pairs[2 * p + (q >> 5)];
    int c = (q & 31) << 2;
    *(uint2*)(comb + (size_t)p * 256 + (q << 2)) =
        *(const uint2*)(h2 + (size_t)node * HH + c);
}

// ---------------------------------------------------------------------------
static inline int cdiv_ll(long long a, int b) { return (int)((a + b - 1) / b); }

extern "C" void kernel_launch(void* const* d_in, const int* in_sizes, int n_in,
                              void* d_out, int out_size, void* d_ws, size_t ws_size,
                              hipStream_t stream) {
    const float* x      = (const float*)d_in[0];
    const int*   ei     = (const int*)d_in[1];       // [2, E]
    const int*   pairs  = (const int*)d_in[2];       // [P, 2]
    const float* W1     = (const float*)d_in[3];
    const float* b1     = (const float*)d_in[4];
    const float* gamma1 = (const float*)d_in[5];
    const float* beta1  = (const float*)d_in[6];
    const float* W2     = (const float*)d_in[7];
    const float* b2     = (const float*)d_in[8];
    const float* gamma2 = (const float*)d_in[9];
    const float* beta2  = (const float*)d_in[10];
    const float* Wc1    = (const float*)d_in[11];
    const float* bc1    = (const float*)d_in[12];
    const float* Wc2    = (const float*)d_in[13];
    const float* bc2    = (const float*)d_in[14];
    const float* Wc3    = (const float*)d_in[15];
    const float* bc3    = (const float*)d_in[16];
    float* out = (float*)d_out;

    const int* src = ei;
    const int* dst = ei + EE;

    // ---- workspace layout (regions reused across phases) -------------------
    char* ws = (char*)d_ws;
    const size_t szNH_f32 = (size_t)NN * HH * 4;     // 25.6 MB
    const size_t szNH_f16 = (size_t)NN * HH * 2;     // 12.8 MB
    // R0: hw | agg  (51.2 MB), later reused as combined [P,256] f16 (51.2 MB)
    float*    hw   = (float*)(ws + 0);
    float*    agg  = (float*)(ws + szNH_f32);
    _Float16* comb = (_Float16*)(ws + 0);
    // R1: x16 | h16b (25.6 MB), later reused as z1 [P,128] f16 (25.6 MB)
    size_t offR1 = 2 * szNH_f32;
    _Float16* x16  = (_Float16*)(ws + offR1);              // layer-1 A; later h2 f16
    _Float16* h16b = (_Float16*)(ws + offR1 + szNH_f16);   // layer-1 BN out (layer-2 A)
    _Float16* z1   = (_Float16*)(ws + offR1);
    // R2: z2 [P,64] f16 (12.8 MB)
    size_t offR2 = offR1 + 2 * szNH_f16;
    _Float16* z2 = (_Float16*)(ws + offR2);
    // misc
    size_t offM = offR2 + (size_t)PP * 64 * 2;
    float* deg    = (float*)(ws + offM);            offM += (size_t)NN * 4;
    float* dinv   = (float*)(ws + offM);            offM += (size_t)NN * 4;
    float* mean   = (float*)(ws + offM);            offM += 512;
    float* rstd   = (float*)(ws + offM);            offM += 512;
    float* colsum = (float*)(ws + offM);            offM += 512;
    float* colsq  = (float*)(ws + offM);            offM += 512;
    _Float16* W1p  = (_Float16*)(ws + offM);        offM += (size_t)128 * 128 * 2;
    _Float16* W2p  = (_Float16*)(ws + offM);        offM += (size_t)128 * 128 * 2;
    _Float16* Wc1p = (_Float16*)(ws + offM);        offM += (size_t)256 * 128 * 2;
    _Float16* Wc2p = (_Float16*)(ws + offM);        offM += (size_t)128 * 64 * 2;
    _Float16* Wc3p = (_Float16*)(ws + offM);        offM += (size_t)64 * OUTPAD * 2;
    (void)ws_size; (void)in_sizes; (void)n_in; (void)out_size;

    const long long NHt = (long long)NN * HH;
    const int bnBlocks = (NN + 63) / 64;

    // ---- degree / normalization -------------------------------------------
    fill_kernel<<<cdiv_ll(NN, 256), 256, 0, stream>>>(deg, 0.0f, NN);
    deg_kernel<<<cdiv_ll(EE, 256), 256, 0, stream>>>(deg, dst, EE);
    dinv_kernel<<<cdiv_ll(NN, 256), 256, 0, stream>>>(dinv, deg, NN);

    // ---- convert / pack ----------------------------------------------------
    f32_to_f16_kernel<<<cdiv_ll(NHt, 256), 256, 0, stream>>>(x, x16, NHt);
    pack_b_kernel<<<cdiv_ll((long long)(128 / 32) * (128 / 16) * 512, 256), 256, 0, stream>>>(
        W1, W1p, 128, 128, 128);
    pack_b_kernel<<<cdiv_ll((long long)(128 / 32) * (128 / 16) * 512, 256), 256, 0, stream>>>(
        W2, W2p, 128, 128, 128);
    pack_b_kernel<<<cdiv_ll((long long)(256 / 32) * (128 / 16) * 512, 256), 256, 0, stream>>>(
        Wc1, Wc1p, 256, 128, 128);
    pack_b_kernel<<<cdiv_ll((long long)(128 / 32) * (64 / 16) * 512, 256), 256, 0, stream>>>(
        Wc2, Wc2p, 128, 64, 64);
    pack_b_kernel<<<cdiv_ll((long long)(64 / 32) * (OUTPAD / 16) * 512, 256), 256, 0, stream>>>(
        Wc3, Wc3p, 64, OUTC, OUTPAD);

    // 8 waves (256 threads) per block; each wave owns one 16-row block
    const int rowTilesN = NN / 16;                    // 3125 row blocks
    const int gemmBlocksN = (rowTilesN + 7) / 8;
    const int rowTilesP = PP / 16;                    // 6250 row blocks
    const int gemmBlocksP = (rowTilesP + 7) / 8;
    const int ldsGcn  = 8 * (128 / 32) * 1024;        // 32 KB
    const int ldsC1   = 8 * (256 / 32) * 1024;        // 64 KB
    const int ldsC2   = 4 * (128 / 32) * 1024;        // 16 KB
    const int ldsC3   = 6 * (64 / 32) * 1024;         // 12 KB

    // ---- GCN layer 1 -------------------------------------------------------
    wmma_gemm_rows_kernel<8, false, false><<<gemmBlocksN, 256, ldsGcn, stream>>>(
        x16, W1p, nullptr, hw, nullptr, NN, 128, HH, HH);
    fill_kernel<<<cdiv_ll(NHt, 256), 256, 0, stream>>>(agg, 0.0f, NHt);
    edge_agg_kernel<<<cdiv_ll((long long)EE * 32, 256), 256, 0, stream>>>(
        hw, agg, src, dst, dinv, EE);
    selfloop_bias_kernel<<<cdiv_ll(NHt, 256), 256, 0, stream>>>(agg, hw, dinv, b1, NHt);
    fill_kernel<<<1, 256, 0, stream>>>(colsum, 0.0f, 128);
    fill_kernel<<<1, 256, 0, stream>>>(colsq, 0.0f, 128);
    bn_partial_kernel<<<bnBlocks, 256, 0, stream>>>(agg, colsum, colsq, NN);
    bn_finalize_kernel<<<1, 128, 0, stream>>>(colsum, colsq, mean, rstd, NN);
    bn_apply_kernel<<<cdiv_ll(NHt, 256), 256, 0, stream>>>(
        agg, gamma1, beta1, mean, rstd, h16b, NHt);

    // ---- GCN layer 2 -------------------------------------------------------
    wmma_gemm_rows_kernel<8, false, false><<<gemmBlocksN, 256, ldsGcn, stream>>>(
        h16b, W2p, nullptr, hw, nullptr, NN, 128, HH, HH);
    fill_kernel<<<cdiv_ll(NHt, 256), 256, 0, stream>>>(agg, 0.0f, NHt);
    edge_agg_kernel<<<cdiv_ll((long long)EE * 32, 256), 256, 0, stream>>>(
        hw, agg, src, dst, dinv, EE);
    selfloop_bias_kernel<<<cdiv_ll(NHt, 256), 256, 0, stream>>>(agg, hw, dinv, b2, NHt);
    fill_kernel<<<1, 256, 0, stream>>>(colsum, 0.0f, 128);
    fill_kernel<<<1, 256, 0, stream>>>(colsq, 0.0f, 128);
    bn_partial_kernel<<<bnBlocks, 256, 0, stream>>>(agg, colsum, colsq, NN);
    bn_finalize_kernel<<<1, 128, 0, stream>>>(colsum, colsq, mean, rstd, NN);
    bn_apply_kernel<<<cdiv_ll(NHt, 256), 256, 0, stream>>>(
        agg, gamma2, beta2, mean, rstd, x16 /* h2 f16, reuses x16 region */, NHt);

    // ---- pair gather + classifier MLP -------------------------------------
    gather_pairs_kernel<<<cdiv_ll((long long)PP * 64, 256), 256, 0, stream>>>(
        x16, pairs, comb, PP);

    wmma_gemm_rows_kernel<8, true, true><<<gemmBlocksP, 256, ldsC1, stream>>>(
        comb, Wc1p, bc1, nullptr, z1, PP, 256, 128, 128);

    wmma_gemm_rows_kernel<4, true, true><<<gemmBlocksP, 256, ldsC2, stream>>>(
        z1, Wc2p, bc2, nullptr, z2, PP, 128, 64, 64);

    wmma_gemm_rows_kernel<6, false, false><<<gemmBlocksP, 256, ldsC3, stream>>>(
        z2, Wc3p, bc3, out, nullptr, PP, 64, OUTC, OUTC);
}
